// MoEFFN_13322988552527
// MI455X (gfx1250) — matrix operand order repaired and checked by reading, not compiled, not run on hardware
//
#include <hip/hip_runtime.h>
#include <hip/hip_bf16.h>

#define D_MODEL   1024
#define DIM_FF    4096
#define N_EXPERTS 8
#define N_TOKENS  8192
#define TOTAL_ROWS (2 * N_TOKENS)   // each token hits exactly TOPK=2 experts

typedef __bf16  bf16_t;
typedef bf16_t  v16bf __attribute__((ext_vector_type(16)));
typedef float   v8f   __attribute__((ext_vector_type(8)));
typedef int     v4i   __attribute__((ext_vector_type(4)));

union Frag32 { uint4 q[2]; v16bf v; };   // 32 bytes: one 16-bit WMMA operand per lane

struct TokInfo { int e0, e1, s0, s1; float w0, w1; };

// gfx1250 async global->LDS copy (ASYNCcnt-tracked); fall back to reg pipeline
#if defined(__AMDGCN__) && __has_builtin(__builtin_amdgcn_global_load_async_to_lds_b128)
  #define USE_ASYNC_LDS 1
#else
  #define USE_ASYNC_LDS 0
#endif

#if USE_ASYNC_LDS
typedef __attribute__((address_space(1))) v4i* gv4ip;   // global int4*
typedef __attribute__((address_space(3))) v4i* lv4ip;   // LDS int4*
__device__ __forceinline__ void async_cp16(const unsigned short* g, unsigned short* l) {
  __builtin_amdgcn_global_load_async_to_lds_b128((gv4ip)g, (lv4ip)l, 0, 0);
}
__device__ __forceinline__ void async_drain() {
#if __has_builtin(__builtin_amdgcn_s_wait_asynccnt)
  __builtin_amdgcn_s_wait_asynccnt(0);
#else
  asm volatile("s_wait_asynccnt 0x0" ::: "memory");
#endif
}
#endif

// fp32 -> bf16 round-to-nearest-even (bit domain)
__device__ __forceinline__ unsigned short f2bf(float f) {
  unsigned u = __float_as_uint(f);
  u += 0x7FFFu + ((u >> 16) & 1u);
  return (unsigned short)(u >> 16);
}

__device__ __forceinline__ float gelu_exact(float v) {
  return 0.5f * v * (1.0f + erff(v * 0.70710678118654752440f));
}

// ---------------------------------------------------------------- gate -----
__global__ __launch_bounds__(256)
void gate_kernel(const float* __restrict__ x, const float* __restrict__ Wg,
                 const float* __restrict__ bg, int* __restrict__ counts,
                 TokInfo* __restrict__ ti, int n) {
  int t = blockIdx.x * blockDim.x + threadIdx.x;
  if (t >= n) return;
  float l[N_EXPERTS];
  #pragma unroll
  for (int e = 0; e < N_EXPERTS; ++e) l[e] = bg[e];
  const float* xr = x + (size_t)t * D_MODEL;
  for (int d = 0; d < D_MODEL; ++d) {
    float xv = xr[d];
    #pragma unroll
    for (int e = 0; e < N_EXPERTS; ++e) l[e] += xv * Wg[d * N_EXPERTS + e];
  }
  float mx = l[0];
  #pragma unroll
  for (int e = 1; e < N_EXPERTS; ++e) mx = fmaxf(mx, l[e]);
  float p[N_EXPERTS], s = 0.f;
  #pragma unroll
  for (int e = 0; e < N_EXPERTS; ++e) { p[e] = __expf(l[e] - mx); s += p[e]; }
  float inv = 1.0f / s;
  // top-2 (stable: lower index wins ties, matches top_k)
  float b0v = -1.f, b1v = -1.f; int b0i = 0, b1i = 0;
  #pragma unroll
  for (int e = 0; e < N_EXPERTS; ++e) {
    float v = p[e] * inv;
    if (v > b0v)      { b1v = b0v; b1i = b0i; b0v = v; b0i = e; }
    else if (v > b1v) { b1v = v; b1i = e; }
  }
  float den = b0v + b1v + 1e-9f;
  TokInfo o;
  o.e0 = b0i; o.e1 = b1i;
  o.s0 = atomicAdd(&counts[b0i], 1);
  o.s1 = atomicAdd(&counts[b1i], 1);
  o.w0 = b0v / den; o.w1 = b1v / den;
  ti[t] = o;
}

__global__ void prefix_kernel(const int* __restrict__ counts, int* __restrict__ base) {
  if (threadIdx.x == 0) {
    int s = 0;
    for (int e = 0; e < N_EXPERTS; ++e) { base[e] = s; s += counts[e]; }
    base[N_EXPERTS] = s;
  }
}

__global__ __launch_bounds__(256)
void scatter_kernel(const TokInfo* __restrict__ ti, const int* __restrict__ base,
                    int* __restrict__ etok, float* __restrict__ ewt, int n) {
  int t = blockIdx.x * blockDim.x + threadIdx.x;
  if (t >= n) return;
  TokInfo v = ti[t];
  int r0 = base[v.e0] + v.s0; etok[r0] = t; ewt[r0] = v.w0;
  int r1 = base[v.e1] + v.s1; etok[r1] = t; ewt[r1] = v.w1;
}

__global__ __launch_bounds__(256)
void cvt_bf16_kernel(const float* __restrict__ x, unsigned short* __restrict__ xb, int n4) {
  int i = blockIdx.x * blockDim.x + threadIdx.x;
  if (i >= n4) return;
  float4 f = reinterpret_cast<const float4*>(x)[i];
  uint2 o;
  o.x = (unsigned)f2bf(f.x) | ((unsigned)f2bf(f.y) << 16);
  o.y = (unsigned)f2bf(f.z) | ((unsigned)f2bf(f.w) << 16);
  reinterpret_cast<uint2*>(xb)[i] = o;
}

// ------------------------------------------------------- grouped WMMA GEMM --
// EPI=1: H = GELU(gather(xb)·W1_e + b1_e)   (bf16 out to workspace)
// EPI=2: out[tok] += w · (H·W2_e + b2_e)    (float atomic scatter-add)
#define BM 128
#define BN 128
#define BK 32
#define LSTR (BK + 8)   // padded LDS row stride (80 B: keeps 16B alignment)

template <int EPI>
__global__ __launch_bounds__(256)
void moe_gemm(const unsigned short* __restrict__ A, const float* __restrict__ W,
              const float* __restrict__ bias, const int* __restrict__ counts,
              const int* __restrict__ rowbase, const int* __restrict__ etok,
              const float* __restrict__ ewt, unsigned short* __restrict__ Hout,
              float* __restrict__ out, int K, int N) {
  const int e  = blockIdx.z;
  const int Ne = counts[e];
  const int m0 = blockIdx.y * BM;
  if (m0 >= Ne) return;                       // uniform early exit (before barriers)
  const int n0 = blockIdx.x * BN;
  const int rb = rowbase[e];
  const float* We = W + (size_t)e * (size_t)K * (size_t)N;

  __shared__ alignas(16) unsigned short As[2][BM * LSTR];
  __shared__ alignas(16) unsigned short Bs[2][BN * LSTR];

  const int tid  = threadIdx.x;
  const int lane = tid & 31, wave = tid >> 5;
  const int wy = (wave >> 2) * 64, wx = (wave & 3) * 32;  // 2x4 wave grid, 64x32 each
  const int half = lane >> 4, lm = lane & 15;

  // A staging: thread -> (row, 16-halfword segment); resolve gathered row once
  const int arow = tid >> 1;
  const int aoff = (tid & 1) * 16;
  int gmr = m0 + arow;
  if (gmr >= Ne) gmr = m0;                    // clamp: stays in-range, result discarded
  const unsigned short* aptr =
      (EPI == 1) ? A + (size_t)etok[rb + gmr] * (size_t)K
                 : A + (size_t)(rb + gmr) * (size_t)K;

  // B staging: thread -> (k-pair, 8-float n segment); packed b32 transpose stores
  const int bkp = tid >> 4;                   // 0..15  -> k = 2*bkp, 2*bkp+1
  const int bn8 = (tid & 15) * 8;             // 0..120
  const float* bp0 = We + (size_t)(2 * bkp) * (size_t)N + n0 + bn8;

  const v8f vzero = {0.f,0.f,0.f,0.f,0.f,0.f,0.f,0.f};
  v8f acc[4][2];
  #pragma unroll
  for (int mi = 0; mi < 4; ++mi)
    #pragma unroll
    for (int ni = 0; ni < 2; ++ni) acc[mi][ni] = vzero;

  uint4  ar0, ar1;                            // A staging regs (non-async path)
  float  fb0[8], fb1[8];                      // B staging regs

  // ---- prologue: stage tile 0 into buffer 0 ----
#if USE_ASYNC_LDS
  async_cp16(aptr + aoff,     &As[0][arow * LSTR + aoff]);
  async_cp16(aptr + aoff + 8, &As[0][arow * LSTR + aoff + 8]);
#else
  {
    const uint4* s = reinterpret_cast<const uint4*>(aptr + aoff);
    ar0 = s[0]; ar1 = s[1];
    uint4* d = reinterpret_cast<uint4*>(&As[0][arow * LSTR + aoff]);
    d[0] = ar0; d[1] = ar1;
  }
#endif
  {
    const float* s0 = bp0;
    const float* s1 = bp0 + N;
    #pragma unroll
    for (int j = 0; j < 8; ++j) { fb0[j] = s0[j]; fb1[j] = s1[j]; }
    #pragma unroll
    for (int j = 0; j < 8; ++j) {
      unsigned v = (unsigned)f2bf(fb0[j]) | ((unsigned)f2bf(fb1[j]) << 16);
      *reinterpret_cast<unsigned*>(&Bs[0][(bn8 + j) * LSTR + 2 * bkp]) = v;
    }
  }
#if USE_ASYNC_LDS
  async_drain();
#endif
  __syncthreads();

  // ---- pipelined mainloop: one barrier per K-step ----
  const int nk = K / BK;
  for (int kt = 0; kt < nk; ++kt) {
    const int cur = kt & 1, nxt = cur ^ 1;
    const int k1 = (kt + 1) * BK;
    const bool hasnext = (k1 < K);

    if (hasnext) {                       // issue next tile's global traffic early
#if USE_ASYNC_LDS
      async_cp16(aptr + k1 + aoff,     &As[nxt][arow * LSTR + aoff]);
      async_cp16(aptr + k1 + aoff + 8, &As[nxt][arow * LSTR + aoff + 8]);
#else
      const uint4* s = reinterpret_cast<const uint4*>(aptr + k1 + aoff);
      ar0 = s[0]; ar1 = s[1];
#endif
      const float* s0 = bp0 + (size_t)k1 * (size_t)N;
      const float* s1 = s0 + N;
      #pragma unroll
      for (int j = 0; j < 8; ++j) { fb0[j] = s0[j]; fb1[j] = s1[j]; }
      if (k1 + BK < K) __builtin_prefetch(s0 + (size_t)BK * (size_t)N, 0, 1);
    }

    // compute on buffer `cur` -- fragments per documented CDNA5 16-bit layouts
    {
      const unsigned short* Ab = As[cur];
      const unsigned short* Bb = Bs[cur];
      Frag32 afr[4], bfr[2];
      #pragma unroll
      for (int mi = 0; mi < 4; ++mi) {     // A elems 0-7: k=half*8+j ; 8-15: k=16+half*8+j
        const uint4* p = reinterpret_cast<const uint4*>(&Ab[(wy + mi*16 + lm) * LSTR + half*8]);
        afr[mi].q[0] = p[0];
        afr[mi].q[1] = p[2];               // +16 halfwords
      }
      #pragma unroll
      for (int ni = 0; ni < 2; ++ni) {     // B elems 0-7: k=half*8+j ; 8-15: k=16+half*8+j
        const uint4* p = reinterpret_cast<const uint4*>(&Bb[(wx + ni*16 + lm) * LSTR + half*8]);
        bfr[ni].q[0] = p[0];
        bfr[ni].q[1] = p[2];
      }
      #pragma unroll
      for (int mi = 0; mi < 4; ++mi)
        #pragma unroll
        for (int ni = 0; ni < 2; ++ni)
          acc[mi][ni] = __builtin_amdgcn_wmma_f32_16x16x32_bf16(
              false, afr[mi].v, false, bfr[ni].v, (short)0, acc[mi][ni], false, false);
    }

    if (hasnext) {                       // commit next tile into buffer `nxt`
#if !USE_ASYNC_LDS
      uint4* d = reinterpret_cast<uint4*>(&As[nxt][arow * LSTR + aoff]);
      d[0] = ar0; d[1] = ar1;
#endif
      #pragma unroll
      for (int j = 0; j < 8; ++j) {
        unsigned v = (unsigned)f2bf(fb0[j]) | ((unsigned)f2bf(fb1[j]) << 16);
        *reinterpret_cast<unsigned*>(&Bs[nxt][(bn8 + j) * LSTR + 2 * bkp]) = v;
      }
#if USE_ASYNC_LDS
      async_drain();
#endif
    }
    __syncthreads();
  }

  // Epilogue. C/D layout: element r -> m = r + 8*half, n = lm (within 16x16 tile)
  #pragma unroll
  for (int mi = 0; mi < 4; ++mi) {
    #pragma unroll
    for (int ni = 0; ni < 2; ++ni) {
      const int ncol = n0 + wx + ni * 16 + lm;
      #pragma unroll
      for (int r = 0; r < 8; ++r) {
        const int gm = m0 + wy + mi * 16 + half * 8 + r;
        if (gm < Ne) {
          float v = acc[mi][ni][r] + bias[e * N + ncol];
          if (EPI == 1) {
            Hout[(size_t)(rb + gm) * (size_t)N + ncol] = f2bf(gelu_exact(v));
          } else {
            const int gr = rb + gm;
            atomicAdd(&out[(size_t)etok[gr] * (size_t)N + ncol], ewt[gr] * v);
          }
        }
      }
    }
  }
}

// ----------------------------------------------------------------- launch --
static inline size_t align256(size_t v) { return (v + 255) & ~(size_t)255; }

extern "C" void kernel_launch(void* const* d_in, const int* in_sizes, int n_in,
                              void* d_out, int out_size, void* d_ws, size_t ws_size,
                              hipStream_t stream) {
  const float* x  = (const float*)d_in[0];
  const float* Wg = (const float*)d_in[1];
  const float* bg = (const float*)d_in[2];
  const float* W1 = (const float*)d_in[3];
  const float* b1 = (const float*)d_in[4];
  const float* W2 = (const float*)d_in[5];
  const float* b2 = (const float*)d_in[6];
  float* out = (float*)d_out;

  char* p = (char*)d_ws;
  int* counts = (int*)p;                       // 8 ints
  int* base   = (int*)(p + 64);                // 9 ints
  size_t off  = 256;
  TokInfo* ti = (TokInfo*)(p + off);           off = align256(off + sizeof(TokInfo) * N_TOKENS);
  int*   etok = (int*)(p + off);               off = align256(off + sizeof(int) * TOTAL_ROWS);
  float* ewt  = (float*)(p + off);             off = align256(off + sizeof(float) * TOTAL_ROWS);
  unsigned short* xb = (unsigned short*)(p + off);
  off = align256(off + (size_t)2 * N_TOKENS * D_MODEL);
  unsigned short* H  = (unsigned short*)(p + off);   // 16384 x 4096 bf16 = 128 MB

  (void)hipMemsetAsync(p, 0, 256, stream);                                   // counts/base
  (void)hipMemsetAsync(d_out, 0, sizeof(float) * (size_t)out_size, stream);  // accumulator

  cvt_bf16_kernel<<<(N_TOKENS * D_MODEL / 4 + 255) / 256, 256, 0, stream>>>(
      x, xb, N_TOKENS * D_MODEL / 4);
  gate_kernel<<<(N_TOKENS + 255) / 256, 256, 0, stream>>>(x, Wg, bg, counts, ti, N_TOKENS);
  prefix_kernel<<<1, 32, 0, stream>>>(counts, base);
  scatter_kernel<<<(N_TOKENS + 255) / 256, 256, 0, stream>>>(ti, base, etok, ewt, N_TOKENS);

  dim3 g1(DIM_FF / BN, N_TOKENS / BM, N_EXPERTS);   // (32, 64, 8), early-exit per expert
  moe_gemm<1><<<g1, 256, 0, stream>>>(xb, W1, b1, counts, base, etok, ewt,
                                      H, nullptr, D_MODEL, DIM_FF);
  dim3 g2(D_MODEL / BN, N_TOKENS / BM, N_EXPERTS);  // (8, 64, 8)
  moe_gemm<2><<<g2, 256, 0, stream>>>(H, W2, b2, counts, base, etok, ewt,
                                      nullptr, out, DIM_FF, D_MODEL);
}
